// CoAttentionMessagePassingNetwork_40750649705201
// MI455X (gfx1250) — matrix operand order repaired and checked by reading, not compiled, not run on hardware
//
#include <hip/hip_runtime.h>

// ---------------------------------------------------------------------------
// CDNA5 (gfx1250) CoAttention MPNN forward.
// All dense GEMMs: v_wmma_f32_16x16x32_f16 (f16 in, f32 accumulate), wave32,
// 8 waves / 256-thread block, 16 rows per wave, EXEC all-ones at every WMMA.
//
// All LDS operand tiles are stored in FRAGMENT-MAJOR layout [tile][lane][16]
// so a lane's fragment is one contiguous v16h -> 2x ds_load_b128 instead of
// 16x ds_load_u16.
//
// Fragment index algebra (from ISA 7.12.2 layouts):
//   A-frag (16x32): m = lane&15,  k = k0 + (e&7) + 8*(lane>>4) + 16*(e>>3)
//   B-frag (32x16): n = lane&15,  k = k0 + (e&7) + 8*(lane>>4) + 16*(e>>3)
// ---------------------------------------------------------------------------

typedef __attribute__((ext_vector_type(16))) _Float16 v16h;
typedef __attribute__((ext_vector_type(8)))  float    v8f;

#define LRELU(x) ((x) > 0.0f ? (x) : 0.01f * (x))

__device__ __forceinline__ v8f wmma_f16(v16h a, v16h b, v8f c) {
  // (neg_a, A, neg_b, B, c_mod, C, reuse_a, reuse_b)
  return __builtin_amdgcn_wmma_f32_16x16x32_f16(false, a, false, b, (short)0, c,
                                                false, false);
}

// A-fragment from f32 global, vectorized: two contiguous 8-float runs.
__device__ __forceinline__ v16h load_a_f32(const float* __restrict__ A, int lda,
                                           int row0, int k0, int lane) {
  int m = lane & 15, hi = lane >> 4;
  const float* base = A + (size_t)(row0 + m) * lda + k0 + 8 * hi;
  const float4* p = reinterpret_cast<const float4*>(base);
  float4 q0 = p[0], q1 = p[1];  // k-off 0..7
  float4 q2 = p[4], q3 = p[5];  // k-off 16..23
  float tmp[16] = {q0.x, q0.y, q0.z, q0.w, q1.x, q1.y, q1.z, q1.w,
                   q2.x, q2.y, q2.z, q2.w, q3.x, q3.y, q3.z, q3.w};
  v16h f;
#pragma unroll
  for (int e = 0; e < 16; ++e) f[e] = (_Float16)tmp[(e & 7) + 8 * (e >> 3)];
  return f;
}

// Stage a [64 x ldw] f32 weight into B-frag layout: tiles kt(KT) x nt(NT).
// dst element i: tile = i>>9, lane = (i>>4)&31, e = i&15.
template <int KT, int NT>
__device__ __forceinline__ void stage_b_frags(const float* __restrict__ W,
                                              int ldw, _Float16* __restrict__ dst,
                                              int tid) {
#pragma unroll 1
  for (int i = tid; i < KT * NT * 512; i += 256) {
    int tile = i >> 9, ln = (i >> 4) & 31, e = i & 15;
    int kt = tile / NT, nt = tile % NT;
    int k = kt * 32 + (e & 7) + 8 * (ln >> 4) + 16 * (e >> 3);
    int n = nt * 16 + (ln & 15);
    dst[i] = (_Float16)W[k * ldw + n];
  }
}

// ===========================================================================
// Kernel 1: pn = node@Wn (f32), kh = node@Wk (f16), vh = node@Wv (f16)
// ===========================================================================
__global__ void kvpn_kernel(const float* __restrict__ node,
                            const float* __restrict__ Wn,
                            const float* __restrict__ Wk,
                            const float* __restrict__ Wv,
                            float* __restrict__ pn,
                            _Float16* __restrict__ kh,
                            _Float16* __restrict__ vh) {
  __shared__ v16h sWf[3][8 * 32];  // kt(2) x nt(4) tiles per weight
  int tid = threadIdx.x;
  stage_b_frags<2, 4>(Wn, 64, (_Float16*)&sWf[0][0], tid);
  stage_b_frags<2, 4>(Wk, 64, (_Float16*)&sWf[1][0], tid);
  stage_b_frags<2, 4>(Wv, 64, (_Float16*)&sWf[2][0], tid);
  __syncthreads();

  int wave = tid >> 5, lane = tid & 31;
  int row0 = blockIdx.x * 128 + wave * 16;
  int n = lane & 15, hf = lane >> 4;
  v16h a0 = load_a_f32(node, 64, row0, 0, lane);
  v16h a1 = load_a_f32(node, 64, row0, 32, lane);
#pragma unroll
  for (int w = 0; w < 3; ++w) {
#pragma unroll
    for (int t = 0; t < 4; ++t) {
      v8f c = {};
      c = wmma_f16(a0, sWf[w][(0 * 4 + t) * 32 + lane], c);
      c = wmma_f16(a1, sWf[w][(1 * 4 + t) * 32 + lane], c);
      if (w == 0) {
#pragma unroll
        for (int r = 0; r < 8; ++r)
          pn[(size_t)(row0 + r + 8 * hf) * 64 + 16 * t + n] = c[r];
      } else {
        _Float16* dst = (w == 1) ? kh : vh;
#pragma unroll
        for (int r = 0; r < 8; ++r)
          dst[(size_t)(row0 + r + 8 * hf) * 64 + 16 * t + n] = (_Float16)c[r];
      }
    }
  }
}

// ===========================================================================
// Kernel 2: fused edge MLP + degree-4 message passing (no atomics).
// Layer-1 activations are written directly into A-frag layout in LDS.
// ===========================================================================
__global__ void edge_mp_kernel(const float* __restrict__ edge,
                               const int* __restrict__ idx,
                               const float* __restrict__ We1,
                               const float* __restrict__ be1,
                               const float* __restrict__ We2,
                               const float* __restrict__ be2,
                               const float* __restrict__ pn,
                               float* __restrict__ inner) {
  __shared__ v16h sW1f[8 * 32], sW2f[8 * 32];
  __shared__ v16h sHf[8][2 * 32];  // per-wave A-frags, kt(2)
  __shared__ float sb1[64], sb2[64];
  int tid = threadIdx.x;
  stage_b_frags<2, 4>(We1, 64, (_Float16*)sW1f, tid);
  stage_b_frags<2, 4>(We2, 64, (_Float16*)sW2f, tid);
  if (tid < 64) { sb1[tid] = be1[tid]; sb2[tid] = be2[tid]; }
  __syncthreads();

  int wave = tid >> 5, lane = tid & 31;
  int e0 = blockIdx.x * 128 + wave * 16;  // first edge row of this wave
  int n = lane & 15, hf = lane >> 4;

  v16h a0 = load_a_f32(edge, 64, e0, 0, lane);
  v16h a1 = load_a_f32(edge, 64, e0, 32, lane);

  // ---- layer 1: write activations straight into A-frag positions
  _Float16* hw = (_Float16*)&sHf[wave][0];
#pragma unroll
  for (int t = 0; t < 4; ++t) {
    v8f c = {};
    c = wmma_f16(a0, sW1f[(0 * 4 + t) * 32 + lane], c);
    c = wmma_f16(a1, sW1f[(1 * 4 + t) * 32 + lane], c);
    float b = sb1[16 * t + n];
    // value at (m = r + 8*hf, col = 16*t + n) -> A-frag (kt, lane_d, e)
    int kt = t >> 1;
    int hi = (n >> 3) & 1;
    int e = (n & 7) + 8 * (t & 1);
#pragma unroll
    for (int r = 0; r < 8; ++r) {
      int m = r + 8 * hf;
      float x = c[r] + b;
      hw[(kt * 32 + (m & 15) + 16 * hi) * 16 + e] = (_Float16)LRELU(x);
    }
  }
  __syncthreads();

  v16h h0 = sHf[wave][0 * 32 + lane];
  v16h h1 = sHf[wave][1 * 32 + lane];

  // ---- layer 2 + fused gather / degree-4 segment reduction
#pragma unroll
  for (int t = 0; t < 4; ++t) {
    v8f c = {};
    c = wmma_f16(h0, sW2f[(0 * 4 + t) * 32 + lane], c);
    c = wmma_f16(h1, sW2f[(1 * 4 + t) * 32 + lane], c);
    float b = sb2[16 * t + n];
    float acc = 0.0f;
#pragma unroll
    for (int r = 0; r < 8; ++r) {
      int eg = e0 + r + 8 * hf;  // global edge id
      float e = c[r] + b;
      e = LRELU(e);
      int src = idx[eg];
      acc += e * pn[(size_t)src * 64 + 16 * t + n];
      if ((r & 3) == 3) {  // finished one node's 4 edges
        inner[(size_t)(eg >> 2) * 64 + 16 * t + n] = acc;
        acc = 0.0f;
      }
    }
  }
}

// ===========================================================================
// Kernel 3: per-graph co-attention (block per graph).
// K/V staged from global directly into fragment layouts; softmax writes its
// results directly into A-frag positions for the msg GEMMs.
// ===========================================================================
__global__ void attn_kernel(const _Float16* __restrict__ k1h,
                            const _Float16* __restrict__ k2h,
                            const _Float16* __restrict__ v1h,
                            const _Float16* __restrict__ v2h,
                            float* __restrict__ msg1, float* __restrict__ msg2,
                            float* __restrict__ a1out, float* __restrict__ a2out) {
  __shared__ v16h sK1f[4 * 32];  // A-frags:  tile = mt*2 + kt
  __shared__ v16h sK2f[4 * 32];  // Bt-frags: tile = kt*2 + nt
  __shared__ v16h sV1f[4 * 32];  // B-frags (K=32): tile = nt
  __shared__ v16h sV2f[4 * 32];
  __shared__ v16h sA1f[2 * 32];  // A-frags (K=32): tile = mt
  __shared__ v16h sA2f[2 * 32];  // A-frags of A2^T
  __shared__ float sT[32 * 32];

  int g = blockIdx.x;
  int tid = threadIdx.x;
  size_t base = (size_t)g * 32 * 64;

  // ---- stage K1 (A-frag), K2 (Bt-frag), V1/V2 (B-frag) from global
  {
    _Float16* dK1 = (_Float16*)sK1f;
    _Float16* dK2 = (_Float16*)sK2f;
    _Float16* dV1 = (_Float16*)sV1f;
    _Float16* dV2 = (_Float16*)sV2f;
#pragma unroll 1
    for (int i = tid; i < 4 * 512; i += 256) {
      int tile = i >> 9, ln = (i >> 4) & 31, e = i & 15;
      int hi = ln >> 4, lo = ln & 15;
      // K1 as A: m = (tile>>1)*16 + lo, k = (tile&1)*32 + (e&7) + 8*hi + 16*(e>>3)
      int mA = (tile >> 1) * 16 + lo;
      int kA = (tile & 1) * 32 + (e & 7) + 8 * hi + 16 * (e >> 3);
      dK1[i] = k1h[base + (size_t)mA * 64 + kA];
      // K2 as B^T: B[k][n] = K2[n][k]; n = (tile&1)*16 + lo, k = (tile>>1)*32 + (e&7) + 8*hi + 16*(e>>3)
      int nB = (tile & 1) * 16 + lo;
      int kB = (tile >> 1) * 32 + (e & 7) + 8 * hi + 16 * (e >> 3);
      dK2[i] = k2h[base + (size_t)nB * 64 + kB];
      // V as B (K=32): k = (e&7) + 8*hi + 16*(e>>3), n = tile*16 + lo
      int kV = (e & 7) + 8 * hi + 16 * (e >> 3);
      int nV = tile * 16 + lo;
      dV1[i] = v1h[base + (size_t)kV * 64 + nV];
      dV2[i] = v2h[base + (size_t)kV * 64 + nV];
    }
  }
  __syncthreads();

  int wave = tid >> 5, lane = tid & 31;
  int n = lane & 15, hf = lane >> 4;

  // ---- logits T = K1 K2^T : 2x2 tiles, K=64 -> waves 0..3
  if (wave < 4) {
    int mt = wave >> 1, nt = wave & 1;
    v8f c = {};
    c = wmma_f16(sK1f[(mt * 2 + 0) * 32 + lane], sK2f[(0 * 2 + nt) * 32 + lane], c);
    c = wmma_f16(sK1f[(mt * 2 + 1) * 32 + lane], sK2f[(1 * 2 + nt) * 32 + lane], c);
#pragma unroll
    for (int r = 0; r < 8; ++r)
      sT[(16 * mt + r + 8 * hf) * 32 + 16 * nt + n] = c[r];
  }
  __syncthreads();

  const float invT = 0.125f;  // 1/sqrt(64)
  _Float16* dA1 = (_Float16*)sA1f;
  _Float16* dA2 = (_Float16*)sA2f;

  // ---- row softmax -> a1 : wave w rows 4w..4w+3, lane = column j
#pragma unroll
  for (int rr = 0; rr < 4; ++rr) {
    int row = wave * 4 + rr;
    int j = lane;
    float v = sT[row * 32 + j];
    float m = v;
    for (int off = 16; off; off >>= 1) m = fmaxf(m, __shfl_xor(m, off, 32));
    float e = __expf((v - m) * invT);
    float z = e;
    for (int off = 16; off; off >>= 1) z += __shfl_xor(z, off, 32);
    float a = e / (z + 1e-8f);
    // A-frag position for A1[m=row][k=j]
    dA1[((row >> 4) * 32 + (row & 15) + 16 * ((j >> 3) & 1)) * 16 +
        (j & 7) + 8 * ((j >> 4) & 1)] = (_Float16)a;
    a1out[(size_t)g * 1024 + row * 32 + j] = a;
  }
  // ---- column softmax -> a2 : wave w cols 4w..4w+3, lane = row i
#pragma unroll
  for (int cc = 0; cc < 4; ++cc) {
    int col = wave * 4 + cc;
    int i = lane;
    float v = sT[i * 32 + col];
    float m = v;
    for (int off = 16; off; off >>= 1) m = fmaxf(m, __shfl_xor(m, off, 32));
    float e = __expf((v - m) * invT);
    float z = e;
    for (int off = 16; off; off >>= 1) z += __shfl_xor(z, off, 32);
    float a = e / (z + 1e-8f);
    // A-frag position for (A2^T)[m=col][k=i]
    dA2[((col >> 4) * 32 + (col & 15) + 16 * ((i >> 3) & 1)) * 16 +
        (i & 7) + 8 * ((i >> 4) & 1)] = (_Float16)a;
    a2out[(size_t)g * 1024 + i * 32 + col] = a;
  }
  __syncthreads();

  // ---- msg GEMMs: 16 tiles (8 msg1 + 8 msg2), 2 per wave, K=32
#pragma unroll
  for (int tt = 0; tt < 2; ++tt) {
    int tile = wave * 2 + tt;
    int which = tile >> 3;
    int mt = (tile >> 2) & 1;
    int nt = tile & 3;
    v8f c = {};
    if (which == 0) {
      c = wmma_f16(sA1f[mt * 32 + lane], sV2f[nt * 32 + lane], c);
#pragma unroll
      for (int r = 0; r < 8; ++r)
        msg1[base + (size_t)(16 * mt + r + 8 * hf) * 64 + 16 * nt + n] = c[r];
    } else {
      c = wmma_f16(sA2f[mt * 32 + lane], sV1f[nt * 32 + lane], c);
#pragma unroll
      for (int r = 0; r < 8; ++r)
        msg2[base + (size_t)(16 * mt + r + 8 * hf) * 64 + 16 * nt + n] = c[r];
    }
  }
}

// ===========================================================================
// Kernel 4: outer = lrelu(msg@Wo+bo); node_out = LN(node_in + inner + outer)
// ===========================================================================
__global__ void post_kernel(const float* __restrict__ msg,
                            const float* __restrict__ node_in,
                            const float* __restrict__ inner,
                            const float* __restrict__ Wo,
                            const float* __restrict__ bo,
                            const float* __restrict__ lng,
                            const float* __restrict__ lnb,
                            float* __restrict__ node_out) {
  __shared__ v16h sWf[8 * 32];
  __shared__ float sbo[64], sg[64], sb[64];
  int tid = threadIdx.x;
  stage_b_frags<2, 4>(Wo, 64, (_Float16*)sWf, tid);
  if (tid < 64) { sbo[tid] = bo[tid]; sg[tid] = lng[tid]; sb[tid] = lnb[tid]; }
  __syncthreads();

  int wave = tid >> 5, lane = tid & 31;
  int row0 = blockIdx.x * 128 + wave * 16;
  int n = lane & 15, hf = lane >> 4;

  v16h a0 = load_a_f32(msg, 64, row0, 0, lane);
  v16h a1 = load_a_f32(msg, 64, row0, 32, lane);

  float x[4][8];
  float s1[8], s2[8];
#pragma unroll
  for (int r = 0; r < 8; ++r) { s1[r] = 0.0f; s2[r] = 0.0f; }

#pragma unroll
  for (int t = 0; t < 4; ++t) {
    v8f c = {};
    c = wmma_f16(a0, sWf[(0 * 4 + t) * 32 + lane], c);
    c = wmma_f16(a1, sWf[(1 * 4 + t) * 32 + lane], c);
    float b = sbo[16 * t + n];
#pragma unroll
    for (int r = 0; r < 8; ++r) {
      size_t off = (size_t)(row0 + r + 8 * hf) * 64 + 16 * t + n;
      float o = c[r] + b;
      o = LRELU(o);
      float xv = node_in[off] + inner[off] + o;
      x[t][r] = xv;
      s1[r] += xv;
      s2[r] += xv * xv;
    }
  }
  // per-row stats across the 16-lane group (xor masks < 16 stay in-group)
#pragma unroll
  for (int r = 0; r < 8; ++r) {
    for (int off = 1; off < 16; off <<= 1) {
      s1[r] += __shfl_xor(s1[r], off, 32);
      s2[r] += __shfl_xor(s2[r], off, 32);
    }
  }
#pragma unroll
  for (int r = 0; r < 8; ++r) {
    float mu = s1[r] * (1.0f / 64.0f);
    float var = s2[r] * (1.0f / 64.0f) - mu * mu;
    float rs = rsqrtf(var + 1e-5f);
    int row = row0 + r + 8 * hf;
#pragma unroll
    for (int t = 0; t < 4; ++t) {
      int col = 16 * t + n;
      node_out[(size_t)row * 64 + col] = (x[t][r] - mu) * rs * sg[col] + sb[col];
    }
  }
}

// ===========================================================================
// Kernel 5: per-graph node sum (32 contiguous rows per graph)
// ===========================================================================
__global__ void gsum_kernel(const float* __restrict__ node, float* __restrict__ gs) {
  int g = blockIdx.x, c = threadIdx.x;  // 64 threads
  float acc = 0.0f;
  for (int p = 0; p < 32; ++p) acc += node[((size_t)g * 32 + p) * 64 + c];
  gs[(size_t)g * 64 + c] = acc;
}

// ===========================================================================
// Kernel 6: readout  out = gs @ ro_W + 32*ro_b   ([1024,64]x[64,256])
// ===========================================================================
__global__ void readout_kernel(const float* __restrict__ gs,
                               const float* __restrict__ roW,
                               const float* __restrict__ rob,
                               float* __restrict__ out) {
  __shared__ v16h sWf[32 * 32];  // kt(2) x nt(16)
  int tid = threadIdx.x;
  stage_b_frags<2, 16>(roW, 256, (_Float16*)sWf, tid);
  __syncthreads();
  int wave = tid >> 5, lane = tid & 31;
  int row0 = blockIdx.x * 128 + wave * 16;
  int n = lane & 15, hf = lane >> 4;
  v16h a0 = load_a_f32(gs, 64, row0, 0, lane);
  v16h a1 = load_a_f32(gs, 64, row0, 32, lane);
#pragma unroll
  for (int t = 0; t < 16; ++t) {
    v8f c = {};
    c = wmma_f16(a0, sWf[(0 * 16 + t) * 32 + lane], c);
    c = wmma_f16(a1, sWf[(1 * 16 + t) * 32 + lane], c);
    float b = 32.0f * rob[16 * t + n];
#pragma unroll
    for (int r = 0; r < 8; ++r)
      out[(size_t)(row0 + r + 8 * hf) * 256 + 16 * t + n] = c[r] + b;
  }
}

// ===========================================================================
// Host orchestration
// ===========================================================================
extern "C" void kernel_launch(void* const* d_in, const int* in_sizes, int n_in,
                              void* d_out, int out_size, void* d_ws, size_t ws_size,
                              hipStream_t stream) {
  (void)in_sizes; (void)n_in; (void)out_size; (void)ws_size;
  const int N = 32768, E = 131072, Bg = 1024, S = 2;

  const float* node1_in = (const float*)d_in[1];
  const float* edge1    = (const float*)d_in[2];
  const int*   idx1     = (const int*)d_in[4];
  const float* node2_in = (const float*)d_in[8];
  const float* edge2    = (const float*)d_in[9];
  const int*   idx2     = (const int*)d_in[11];
  const float* mp_Wn  = (const float*)d_in[14];
  const float* mp_We1 = (const float*)d_in[15];
  const float* mp_be1 = (const float*)d_in[16];
  const float* mp_We2 = (const float*)d_in[17];
  const float* mp_be2 = (const float*)d_in[18];
  const float* ca_Wk  = (const float*)d_in[19];
  const float* ca_Wv  = (const float*)d_in[20];
  const float* ca_Wo  = (const float*)d_in[21];
  const float* ca_bo  = (const float*)d_in[22];
  const float* ln_g   = (const float*)d_in[23];
  const float* ln_b   = (const float*)d_in[24];
  const float* ro_W   = (const float*)d_in[25];
  const float* ro_b   = (const float*)d_in[26];

  float* out_g1 = (float*)d_out;            // [1024,256]
  float* out_g2 = out_g1 + Bg * 256;        // [1024,256]
  float* out_a1 = out_g2 + Bg * 256;        // [E_OUT,1]
  float* out_a2 = out_a1 + (size_t)N * 32;  // [E_OUT,1]

  char* wsp = (char*)d_ws;
  auto alloc = [&](size_t bytes) -> void* {
    void* p = (void*)wsp;
    wsp += (bytes + 255) & ~(size_t)255;
    return p;
  };
  float*    nodebuf1 = (float*)alloc((size_t)N * 64 * 4);
  float*    nodebuf2 = (float*)alloc((size_t)N * 64 * 4);
  float*    pn1      = (float*)alloc((size_t)N * 64 * 4);  // reused as msg1
  float*    pn2      = (float*)alloc((size_t)N * 64 * 4);  // reused as msg2
  float*    inner1   = (float*)alloc((size_t)N * 64 * 4);
  float*    inner2   = (float*)alloc((size_t)N * 64 * 4);
  _Float16* k1h      = (_Float16*)alloc((size_t)N * 64 * 2);
  _Float16* k2h      = (_Float16*)alloc((size_t)N * 64 * 2);
  _Float16* v1h      = (_Float16*)alloc((size_t)N * 64 * 2);
  _Float16* v2h      = (_Float16*)alloc((size_t)N * 64 * 2);
  float*    gs1      = (float*)alloc((size_t)Bg * 64 * 4);
  float*    gs2      = (float*)alloc((size_t)Bg * 64 * 4);

  for (int s = 0; s < S; ++s) {
    const float* n1 = s ? nodebuf1 : node1_in;
    const float* n2 = s ? nodebuf2 : node2_in;
    const float* Wn  = mp_Wn  + s * 4096;
    const float* We1 = mp_We1 + s * 4096;
    const float* be1 = mp_be1 + s * 64;
    const float* We2 = mp_We2 + s * 4096;
    const float* be2 = mp_be2 + s * 64;
    const float* Wk  = ca_Wk  + s * 4096;
    const float* Wv  = ca_Wv  + s * 4096;
    const float* Wo  = ca_Wo  + s * 4096;
    const float* bo  = ca_bo  + s * 64;
    const float* lg  = ln_g   + s * 64;
    const float* lb  = ln_b   + s * 64;

    kvpn_kernel<<<N / 128, 256, 0, stream>>>(n1, Wn, Wk, Wv, pn1, k1h, v1h);
    kvpn_kernel<<<N / 128, 256, 0, stream>>>(n2, Wn, Wk, Wv, pn2, k2h, v2h);

    edge_mp_kernel<<<E / 128, 256, 0, stream>>>(edge1, idx1, We1, be1, We2, be2,
                                                pn1, inner1);
    edge_mp_kernel<<<E / 128, 256, 0, stream>>>(edge2, idx2, We1, be1, We2, be2,
                                                pn2, inner2);

    // pn buffers are dead now; attention writes msg into them
    attn_kernel<<<Bg, 256, 0, stream>>>(k1h, k2h, v1h, v2h, pn1, pn2,
                                        out_a1, out_a2);

    post_kernel<<<N / 128, 256, 0, stream>>>(pn1, n1, inner1, Wo, bo, lg, lb,
                                             nodebuf1);
    post_kernel<<<N / 128, 256, 0, stream>>>(pn2, n2, inner2, Wo, bo, lg, lb,
                                             nodebuf2);
  }

  gsum_kernel<<<Bg, 64, 0, stream>>>(nodebuf1, gs1);
  gsum_kernel<<<Bg, 64, 0, stream>>>(nodebuf2, gs2);
  readout_kernel<<<Bg / 128, 256, 0, stream>>>(gs1, ro_W, ro_b, out_g1);
  readout_kernel<<<Bg / 128, 256, 0, stream>>>(gs2, ro_W, ro_b, out_g2);
}
